// T5Voice_45234595561601
// MI455X (gfx1250) — matrix operand order repaired and checked by reading, not compiled, not run on hardware
//
#include <hip/hip_runtime.h>
#include <math.h>

// ---------------------------------------------------------------------------
// T5 encoder self-attention on gfx1250 (MI455X), wave32 + WMMA bf16 + TDM.
// ---------------------------------------------------------------------------

typedef __attribute__((ext_vector_type(16))) __bf16 v16bf;
typedef __attribute__((ext_vector_type(8)))  float  v8f;
typedef __attribute__((ext_vector_type(4)))  unsigned int v4u;
typedef __attribute__((ext_vector_type(4)))  int    v4i;
typedef __attribute__((ext_vector_type(8)))  int    v8i;

union FragBF {
    v16bf v;
    unsigned int u32[8];
    unsigned short u16[16];
};
union FragF {
    v8f v;
    float f[8];
};

#define WMMA_BF16(C, A, B)                                                     \
    (C) = __builtin_amdgcn_wmma_f32_16x16x32_bf16(false, (A), false, (B),      \
                                                  (short)0, (C), false, false)

#if __has_builtin(__builtin_amdgcn_tensor_load_to_lds) &&                      \
    __has_builtin(__builtin_amdgcn_s_wait_tensorcnt)
#define HAS_TDM 1
#else
#define HAS_TDM 0
#endif

__device__ __forceinline__ unsigned short f2bf(float x) {
    unsigned int u = __float_as_uint(x);
    unsigned int r = (u + 0x7FFFu + ((u >> 16) & 1u)) >> 16;
    return (unsigned short)r;
}

#if HAS_TDM
// Set bits [lo, lo+width) of a dword array to val (ISA D# field packing).
__device__ __forceinline__ void dset(unsigned int* a, int lo, int width,
                                     unsigned long long val) {
#pragma unroll
    for (int i = 0; i < width; ++i) {
        if ((val >> i) & 1ull) a[(lo + i) >> 5] |= 1u << ((lo + i) & 31);
    }
}

// Issue one TDM 2D tile load: tensor [tensor_h][tensor_w] of 2-byte elems,
// tile [tile_h][tile_w] starting at gptr, packed row-major into LDS at lds.
// ISA 08_async_tensor §8: group0 = count/lds_addr/global_addr/type,
// group1 = data_size/tensor dims/tile dims/stride. Groups 2,3 unused (2D).
__device__ __forceinline__ void tdm_load_2d(const void* gptr, void* lds,
                                            unsigned tensor_w,
                                            unsigned tensor_h,
                                            unsigned tile_w,
                                            unsigned tile_h) {
    unsigned int g0[4] = {0u, 0u, 0u, 0u};
    unsigned int g1[8] = {0u, 0u, 0u, 0u, 0u, 0u, 0u, 0u};
    const unsigned long long ga = (unsigned long long)gptr;
    // Generic LDS pointer: low 32 bits are the LDS byte offset (ISA 10.2).
    const unsigned int laddr = (unsigned int)(unsigned long long)lds;

    g0[0] = 1u;                                   // count = 1 valid descriptor
    g0[1] = laddr;                                // lds_addr [63:32]
    g0[2] = (unsigned int)ga;                     // global_addr [95:64]
    g0[3] = (unsigned int)((ga >> 32) & 0x01FFFFFFu) | (2u << 30); // +type=2

    g1[0] = 1u << 16;                             // data_size = 1 (2 bytes)
    dset(g1, 48, 32, tensor_w);                   // tensor_dim0
    dset(g1, 80, 32, tensor_h);                   // tensor_dim1
    dset(g1, 112, 16, tile_w);                    // tile_dim0
    dset(g1, 128, 16, tile_h);                    // tile_dim1
    dset(g1, 160, 48, tensor_w);                  // tensor_dim0_stride

    v4u G0 = {g0[0], g0[1], g0[2], g0[3]};
    v8i G1 = {(int)g1[0], (int)g1[1], (int)g1[2], (int)g1[3],
              (int)g1[4], (int)g1[5], (int)g1[6], (int)g1[7]};
    v4i Z4 = {0, 0, 0, 0};
#if defined(__clang_major__) && (__clang_major__ >= 23)
    v8i Z8 = {0, 0, 0, 0, 0, 0, 0, 0};
    __builtin_amdgcn_tensor_load_to_lds(G0, G1, Z4, Z4, Z8, 0);
#else
    __builtin_amdgcn_tensor_load_to_lds(G0, G1, Z4, Z4, 0);
#endif
}
#endif  // HAS_TDM

// A-matrix fragment (16x32 bf16) from LDS, rows mbase..mbase+15.
// Layout: lanes 0-15 hold K=0..15; lanes 16-31 K=16..31 (matches B frag).
__device__ __forceinline__ void load_a_frag(const unsigned short* lds,
                                            int row_stride, int mbase,
                                            int lane, FragBF& f) {
    const int h = lane >> 4;
    const int m = mbase + (lane & 15);
    const unsigned short* p = lds + m * row_stride;
#pragma unroll
    for (int v = 0; v < 8; ++v) {
        const int kb = 16 * h + 2 * v;
        f.u32[v] = *(const unsigned int*)(p + kb);
    }
}

// B-matrix fragment (32x16 bf16) from a TRANSPOSED LDS tile Bt[n][k].
// Layout used here (same as A frag): lane n%16 holds column n; VGPR v holds K = 16*(lane/16)+2v, +1.
__device__ __forceinline__ void load_bt_frag(const unsigned short* lds,
                                             int row_stride, int nbase,
                                             int lane, FragBF& f) {
    const int h = lane >> 4;
    const int n = nbase + (lane & 15);
    const unsigned short* p = lds + n * row_stride + 16 * h;
#pragma unroll
    for (int v = 0; v < 8; ++v) {
        f.u32[v] = *(const unsigned int*)(p + 2 * v);
    }
}

// ---------------------------------------------------------------------------
// Kernel 1: T5 relative-position bias table: bias[h][rp + 1023], rp = k - q.
// ---------------------------------------------------------------------------
__global__ __launch_bounds__(256) void bias_table_kernel(
    const float* __restrict__ rel_emb, float* __restrict__ bias) {
    const int i = blockIdx.x * blockDim.x + threadIdx.x;
    if (i >= 16 * 2047) return;
    const int h = i / 2047;
    const int rp = (i % 2047) - 1023;  // memory - context
    int bucket = (rp > 0) ? 16 : 0;
    const int a = rp < 0 ? -rp : rp;
    if (a < 8) {
        bucket += a;
    } else {
        int large = 8 + (int)(logf((float)a * (1.0f / 8.0f)) *
                              (8.0f / logf(16.0f)));
        bucket += (large < 15) ? large : 15;
    }
    bias[i] = rel_emb[bucket * 16 + h];
}

// ---------------------------------------------------------------------------
// Kernel 2: QKV projection. X[4096,1024] fp32 x W[1024,1024] fp32 -> bf16
// outputs in [B,H,S,64] layout. Block tile 64x128, 8 waves, wave tile 32x32.
// (f32->bf16 conversion happens in staging, so this path stays manual.)
// ---------------------------------------------------------------------------
__global__ __launch_bounds__(256) void gemm_qkv_kernel(
    const float* __restrict__ X, const float* __restrict__ Wq,
    const float* __restrict__ Wk, const float* __restrict__ Wv,
    unsigned short* __restrict__ Qb, unsigned short* __restrict__ Kb,
    unsigned short* __restrict__ Vb) {
    __shared__ unsigned short As[64 * 34];
    __shared__ unsigned short Bts[128 * 34];

    const int tid = threadIdx.x;
    const int lane = tid & 31;
    const int wv = tid >> 5;
    const int wr = wv >> 2;
    const int wc = wv & 3;

    const int cb = blockIdx.x;
    const int mat = cb >> 3;
    const int cbl = cb & 7;
    const float* W = (mat == 0) ? Wq : ((mat == 1) ? Wk : Wv);
    unsigned short* Ob = (mat == 0) ? Qb : ((mat == 1) ? Kb : Vb);
    const int n0 = cbl * 128;
    const int m0 = blockIdx.y * 64;

    FragF c[2][2];
#pragma unroll
    for (int mi = 0; mi < 2; ++mi)
#pragma unroll
        for (int ni = 0; ni < 2; ++ni)
#pragma unroll
            for (int j = 0; j < 8; ++j) c[mi][ni].f[j] = 0.0f;

    for (int kt = 0; kt < 32; ++kt) {
        const int k0 = kt * 32;
#pragma unroll
        for (int i = 0; i < 8; ++i) {
            const int idx = tid + i * 256;
            const int r = idx >> 5, k = idx & 31;
            As[r * 34 + k] = f2bf(X[(size_t)(m0 + r) * 1024 + k0 + k]);
        }
#pragma unroll
        for (int i = 0; i < 16; ++i) {
            const int idx = tid + i * 256;
            const int k = idx >> 7, n = idx & 127;
            Bts[n * 34 + k] = f2bf(W[(size_t)(k0 + k) * 1024 + n0 + n]);
        }
        __syncthreads();

        FragBF a[2], bfr[2];
        load_a_frag(As, 34, 32 * wr, lane, a[0]);
        load_a_frag(As, 34, 32 * wr + 16, lane, a[1]);
        load_bt_frag(Bts, 34, 32 * wc, lane, bfr[0]);
        load_bt_frag(Bts, 34, 32 * wc + 16, lane, bfr[1]);

#pragma unroll
        for (int mi = 0; mi < 2; ++mi)
#pragma unroll
            for (int ni = 0; ni < 2; ++ni)
                WMMA_BF16(c[mi][ni].v, a[mi].v, bfr[ni].v);
        __syncthreads();
    }

    const int nlo = lane & 15;
    const int hh = lane >> 4;
#pragma unroll
    for (int mi = 0; mi < 2; ++mi) {
#pragma unroll
        for (int ni = 0; ni < 2; ++ni) {
#pragma unroll
            for (int v = 0; v < 8; ++v) {
                const int r = m0 + 32 * wr + 16 * mi + v + 8 * hh;
                const int col = cbl * 128 + 32 * wc + 16 * ni + nlo;
                const int bb = r >> 10, s = r & 1023;
                const int hd = col >> 6, d = col & 63;
                Ob[(((size_t)bb * 16 + hd) * 1024 + s) * 64 + d] =
                    f2bf(c[mi][ni].f[v]);
            }
        }
    }
}

// ---------------------------------------------------------------------------
// Kernel 3: fused attention (flash style). grid = (S/64, H, B), 128 threads
// (4 waves), wave owns 16 q-rows. Q/K tiles staged by the Tensor Data Mover
// (wave 0 issues, waits TENSORcnt, then workgroup barrier). V staged
// transposed manually; P re-swizzled via per-wave LDS tile.
// ---------------------------------------------------------------------------
__global__ __launch_bounds__(128) void attn_kernel(
    const unsigned short* __restrict__ Q, const unsigned short* __restrict__ K,
    const unsigned short* __restrict__ V, const float* __restrict__ bias,
    unsigned short* __restrict__ ctx) {
    __shared__ unsigned short Qs[64 * 64];     // TDM-filled, packed rows
    __shared__ unsigned short Ks[64 * 64];     // TDM-filled, packed rows
    __shared__ unsigned short Vts[64 * 68];    // V transposed [d][sk]
    __shared__ unsigned short Ps[4][16 * 64];  // per-wave P tile

    const int tid = threadIdx.x;
    const int lane = tid & 31;
    const int wv = tid >> 5;
    const int qb = blockIdx.x * 64;
    const int h = blockIdx.y;
    const int b = blockIdx.z;

    const size_t head = (size_t)(b * 16 + h) * 1024 * 64;
    const unsigned short* Qh = Q + head;
    const unsigned short* Kh = K + head;
    const unsigned short* Vh = V + head;
    const float* biash = bias + h * 2047;

    // ---- stage Q block (64x64 bf16) ----
#if HAS_TDM
    if (wv == 0) {
        tdm_load_2d(Qh + (size_t)qb * 64, Qs, 64u, 1024u, 64u, 64u);
        __builtin_amdgcn_s_wait_tensorcnt(0);
    }
#else
#pragma unroll
    for (int i = 0; i < 16; ++i) {
        const int idx = tid + i * 128;
        const int r = idx >> 5, k2 = idx & 31;
        *(unsigned int*)(Qs + r * 64 + 2 * k2) =
            *(const unsigned int*)(Qh + (qb + r) * 64 + 2 * k2);
    }
#endif
    __syncthreads();

    FragBF qf[2];
    load_a_frag(Qs, 64, 16 * wv, lane, qf[0]);       // d = 0..31
    load_a_frag(Qs + 32, 64, 16 * wv, lane, qf[1]);  // d = 32..63

    float mrow[8], lrow[8];
    FragF o[4];
#pragma unroll
    for (int j = 0; j < 8; ++j) {
        mrow[j] = -INFINITY;
        lrow[j] = 0.0f;
    }
#pragma unroll
    for (int nt = 0; nt < 4; ++nt)
#pragma unroll
        for (int j = 0; j < 8; ++j) o[nt].f[j] = 0.0f;

    const int nlo = lane & 15;
    const int hh = lane >> 4;
    const int qrow_base = qb + 16 * wv + 8 * hh;

    for (int kb = 0; kb < 1024; kb += 64) {
        // ---- stage K (TDM) and V-transposed (manual) ----
#if HAS_TDM
        if (wv == 0) {
            tdm_load_2d(Kh + (size_t)kb * 64, Ks, 64u, 1024u, 64u, 64u);
        }
#else
#pragma unroll
        for (int i = 0; i < 16; ++i) {
            const int idx = tid + i * 128;
            const int r = idx >> 5, k2 = idx & 31;
            *(unsigned int*)(Ks + r * 64 + 2 * k2) =
                *(const unsigned int*)(Kh + (kb + r) * 64 + 2 * k2);
        }
#endif
#pragma unroll
        for (int i = 0; i < 32; ++i) {
            const int idx = tid + i * 128;
            const int r = idx >> 6, d = idx & 63;
            Vts[d * 68 + r] = Vh[(kb + r) * 64 + d];
        }
        if (kb + 64 < 1024) {  // prefetch next tiles toward L2
            __builtin_prefetch(Kh + (size_t)(kb + 64) * 64, 0, 0);
            __builtin_prefetch(Vh + (size_t)(kb + 64) * 64, 0, 0);
        }
#if HAS_TDM
        if (wv == 0) __builtin_amdgcn_s_wait_tensorcnt(0);
#endif
        __syncthreads();

        // ---- scores: 16 x 64 per wave (K row-major == transposed B) ----
        FragF c[4];
#pragma unroll
        for (int nt = 0; nt < 4; ++nt) {
#pragma unroll
            for (int j = 0; j < 8; ++j) c[nt].f[j] = 0.0f;
            FragBF b0, b1;
            load_bt_frag(Ks, 64, nt * 16, lane, b0);
            load_bt_frag(Ks + 32, 64, nt * 16, lane, b1);
            WMMA_BF16(c[nt].v, qf[0].v, b0.v);
            WMMA_BF16(c[nt].v, qf[1].v, b1.v);
        }

        // ---- T5 position bias ----
#pragma unroll
        for (int nt = 0; nt < 4; ++nt) {
            const int kcol = kb + nt * 16 + nlo;
#pragma unroll
            for (int j = 0; j < 8; ++j)
                c[nt].f[j] += biash[kcol - (qrow_base + j) + 1023];
        }

        // ---- online softmax (row m spans 16 lanes of one half-wave) ----
        unsigned short* Pw = Ps[wv];
#pragma unroll
        for (int j = 0; j < 8; ++j) {
            float m_ = c[0].f[j];
#pragma unroll
            for (int nt = 1; nt < 4; ++nt) m_ = fmaxf(m_, c[nt].f[j]);
#pragma unroll
            for (int off = 1; off < 16; off <<= 1)
                m_ = fmaxf(m_, __shfl_xor(m_, off, 32));
            const float mn = fmaxf(mrow[j], m_);
            const float sc = __expf(mrow[j] - mn);
            mrow[j] = mn;
            float rs = 0.0f;
#pragma unroll
            for (int nt = 0; nt < 4; ++nt) {
                const float p = __expf(c[nt].f[j] - mn);
                c[nt].f[j] = p;
                rs += p;
            }
#pragma unroll
            for (int off = 1; off < 16; off <<= 1)
                rs += __shfl_xor(rs, off, 32);
            lrow[j] = lrow[j] * sc + rs;
#pragma unroll
            for (int nt = 0; nt < 4; ++nt) o[nt].f[j] *= sc;
#pragma unroll
            for (int nt = 0; nt < 4; ++nt)
                Pw[(j + 8 * hh) * 64 + nt * 16 + nlo] = f2bf(c[nt].f[j]);
        }

        // ---- O += P @ V ----
        FragBF pa0, pa1;
        load_a_frag(Pw, 64, 0, lane, pa0);
        load_a_frag(Pw + 32, 64, 0, lane, pa1);
#pragma unroll
        for (int nt = 0; nt < 4; ++nt) {
            FragBF v0, v1;
            load_bt_frag(Vts, 68, nt * 16, lane, v0);
            load_bt_frag(Vts + 32, 68, nt * 16, lane, v1);
            WMMA_BF16(o[nt].v, pa0.v, v0.v);
            WMMA_BF16(o[nt].v, pa1.v, v1.v);
        }
        __syncthreads();
    }

    // ---- normalize and write ctx[b][s][h*64+d] bf16 ----
#pragma unroll
    for (int nt = 0; nt < 4; ++nt) {
#pragma unroll
        for (int j = 0; j < 8; ++j) {
            const int s = qrow_base + j;
            const int col = h * 64 + nt * 16 + nlo;
            ctx[((size_t)b * 1024 + s) * 1024 + col] =
                f2bf(o[nt].f[j] / lrow[j]);
        }
    }
}

// ---------------------------------------------------------------------------
// Kernel 4: output projection ctx(bf16)[4096,1024] @ Wo(f32->bf16) -> f32.
// A tile (already bf16) staged via TDM; Wo tile converted manually.
// ---------------------------------------------------------------------------
__global__ __launch_bounds__(256) void gemm_out_kernel(
    const unsigned short* __restrict__ ctx, const float* __restrict__ Wo,
    float* __restrict__ out) {
    __shared__ unsigned short As[64 * 32];    // TDM-filled, packed rows
    __shared__ unsigned short Bts[128 * 34];  // transposed [n][k]

    const int tid = threadIdx.x;
    const int lane = tid & 31;
    const int wv = tid >> 5;
    const int wr = wv >> 2;
    const int wc = wv & 3;
    const int n0 = blockIdx.x * 128;
    const int m0 = blockIdx.y * 64;

    FragF c[2][2];
#pragma unroll
    for (int mi = 0; mi < 2; ++mi)
#pragma unroll
        for (int ni = 0; ni < 2; ++ni)
#pragma unroll
            for (int j = 0; j < 8; ++j) c[mi][ni].f[j] = 0.0f;

    for (int kt = 0; kt < 32; ++kt) {
        const int k0 = kt * 32;
#if HAS_TDM
        if (wv == 0) {
            tdm_load_2d(ctx + (size_t)m0 * 1024 + k0, As, 1024u, 4096u, 32u,
                        64u);
        }
#else
#pragma unroll
        for (int i = 0; i < 8; ++i) {
            const int idx = tid + i * 256;
            const int r = idx >> 5, k = idx & 31;
            As[r * 32 + k] = ctx[(size_t)(m0 + r) * 1024 + k0 + k];
        }
#endif
#pragma unroll
        for (int i = 0; i < 16; ++i) {
            const int idx = tid + i * 256;
            const int k = idx >> 7, n = idx & 127;
            Bts[n * 34 + k] = f2bf(Wo[(size_t)(k0 + k) * 1024 + n0 + n]);
        }
#if HAS_TDM
        if (wv == 0) __builtin_amdgcn_s_wait_tensorcnt(0);
#endif
        __syncthreads();

        FragBF a[2], bfr[2];
        load_a_frag(As, 32, 32 * wr, lane, a[0]);
        load_a_frag(As, 32, 32 * wr + 16, lane, a[1]);
        load_bt_frag(Bts, 34, 32 * wc, lane, bfr[0]);
        load_bt_frag(Bts, 34, 32 * wc + 16, lane, bfr[1]);

#pragma unroll
        for (int mi = 0; mi < 2; ++mi)
#pragma unroll
            for (int ni = 0; ni < 2; ++ni)
                WMMA_BF16(c[mi][ni].v, a[mi].v, bfr[ni].v);
        __syncthreads();
    }

    const int nlo = lane & 15;
    const int hh = lane >> 4;
#pragma unroll
    for (int mi = 0; mi < 2; ++mi)
#pragma unroll
        for (int ni = 0; ni < 2; ++ni)
#pragma unroll
            for (int v = 0; v < 8; ++v) {
                const int r = m0 + 32 * wr + 16 * mi + v + 8 * hh;
                const int col = n0 + 32 * wc + 16 * ni + nlo;
                out[(size_t)r * 1024 + col] = c[mi][ni].f[v];
            }
}

// ---------------------------------------------------------------------------
// Launch
// ---------------------------------------------------------------------------
extern "C" void kernel_launch(void* const* d_in, const int* in_sizes, int n_in,
                              void* d_out, int out_size, void* d_ws,
                              size_t ws_size, hipStream_t stream) {
    const float* X   = (const float*)d_in[0];  // [4,1024,1024]
    const float* Wq  = (const float*)d_in[1];  // [1024,1024]
    const float* Wk  = (const float*)d_in[2];
    const float* Wv  = (const float*)d_in[3];
    const float* Wo  = (const float*)d_in[4];
    const float* rel = (const float*)d_in[5];  // [32,16]
    float* out = (float*)d_out;

    char* w = (char*)d_ws;
    const size_t SZ_HEAD =
        (size_t)4 * 16 * 1024 * 64 * sizeof(unsigned short);  // 8 MB
    unsigned short* Qb  = (unsigned short*)(w);
    unsigned short* Kb  = (unsigned short*)(w + SZ_HEAD);
    unsigned short* Vb  = (unsigned short*)(w + 2 * SZ_HEAD);
    unsigned short* ctx = (unsigned short*)(w + 3 * SZ_HEAD);
    float* bias = (float*)(w + 4 * SZ_HEAD);  // 16*2047 f32

    bias_table_kernel<<<(16 * 2047 + 255) / 256, 256, 0, stream>>>(rel, bias);

    dim3 g1(24, 64);
    gemm_qkv_kernel<<<g1, 256, 0, stream>>>(X, Wq, Wk, Wv, Qb, Kb, Vb);

    dim3 g2(16, 16, 4);
    attn_kernel<<<g2, 128, 0, stream>>>(Qb, Kb, Vb, bias, ctx);

    dim3 g3(8, 64);
    gemm_out_kernel<<<g3, 256, 0, stream>>>(ctx, Wo, out);
}
